// AdaptivePointCNP_23321672417655
// MI455X (gfx1250) — matrix operand order repaired and checked by reading, not compiled, not run on hardware
//
#include <hip/hip_runtime.h>
#include <math.h>

// ---------------- problem constants (from reference setup_inputs) ------------
#define B_    2
#define C_    1024
#define T_    1024
#define N_    2048           // C + T
#define BN_   (B_*N_)        // 4096 points
#define KT_   300            // k_theta
#define KC_   5              // k_cnn
#define R5_   (BN_*KC_)      // 20480 rows for cnn weightnet
#define RT_   (BN_*KT_)      // 1228800 rows for theta weightnet
#define EPS_  1e-5f

typedef __attribute__((ext_vector_type(16))) _Float16 v16h;
typedef __attribute__((ext_vector_type(8)))  float    v8f;

__device__ __forceinline__ float siluf(float x) { return x / (1.f + __expf(-x)); }

// ---------------------------------------------------------------------------
// Utility: zero a float buffer
// ---------------------------------------------------------------------------
__global__ void zero_k(float* __restrict__ p, int n) {
    int i = blockIdx.x * blockDim.x + threadIdx.x;
    if (i < n) p[i] = 0.f;
}

// ---------------------------------------------------------------------------
// Transpose small weight matrix: src (K x N row-major) -> dst (N x K row-major)
// Makes the WMMA B-operand fragment contiguous per lane.
// ---------------------------------------------------------------------------
__global__ void transpose_k(const float* __restrict__ src, float* __restrict__ dst,
                            int K, int Nn) {
    int i = blockIdx.x * blockDim.x + threadIdx.x;
    if (i >= K * Nn) return;
    int k = i / Nn, n = i % Nn;
    dst[(size_t)n * K + k] = src[i];
}

// ---------------------------------------------------------------------------
// Build concatenated coords (B, N, 2)
// ---------------------------------------------------------------------------
__global__ void build_coords_k(const float* __restrict__ cc, const float* __restrict__ tc,
                               float* __restrict__ coords) {
    int i = blockIdx.x * blockDim.x + threadIdx.x;
    if (i >= BN_) return;
    int b = i / N_, n = i % N_;
    if (n < C_) {
        coords[i*2+0] = cc[(b*C_ + n)*2 + 0];
        coords[i*2+1] = cc[(b*C_ + n)*2 + 1];
    } else {
        coords[i*2+0] = tc[(b*T_ + (n-C_))*2 + 0];
        coords[i*2+1] = tc[(b*T_ + (n-C_))*2 + 1];
    }
}

// ---------------------------------------------------------------------------
// Per-point 300-NN via full in-LDS bitonic sort of all 2048 candidates.
// One workgroup (256 threads) per query point; 32KB LDS working set
// (well inside 320KB/WGP).
// ---------------------------------------------------------------------------
__global__ void __launch_bounds__(256) topk_k(const float* __restrict__ coords,
                                              int* __restrict__ nbr) {
    __shared__ float skey[N_];
    __shared__ int   sval[N_];
    __shared__ float scx[N_];
    __shared__ float scy[N_];
    int p = blockIdx.x;
    int b = p / N_, n = p % N_;
    int tid = threadIdx.x;
    for (int m = tid; m < N_; m += 256) {
        scx[m] = coords[(b*N_ + m)*2 + 0];
        scy[m] = coords[(b*N_ + m)*2 + 1];
    }
    __syncthreads();
    float qx = scx[n], qy = scy[n];
    for (int m = tid; m < N_; m += 256) {
        float dx = scx[m] - qx, dy = scy[m] - qy;
        skey[m] = dx*dx + dy*dy;
        sval[m] = m;
    }
    __syncthreads();
    for (int k = 2; k <= N_; k <<= 1) {
        for (int j = k >> 1; j > 0; j >>= 1) {
            #pragma unroll
            for (int s = 0; s < N_/256; ++s) {
                int i = tid + s*256;
                int ixj = i ^ j;
                if (ixj > i) {
                    bool up = ((i & k) == 0);
                    float ki = skey[i], kj = skey[ixj];
                    if ((ki > kj) == up) {
                        skey[i] = kj; skey[ixj] = ki;
                        int t = sval[i]; sval[i] = sval[ixj]; sval[ixj] = t;
                    }
                }
            }
            __syncthreads();
        }
    }
    for (int j = tid; j < KT_; j += 256) nbr[p*KT_ + j] = sval[j];
}

// ---------------------------------------------------------------------------
// Theta weightnet stats (recompute passes, avoids 157MB intermediates).
// stats layout: [0..31]=sum, [32..63]=sumsq
// ---------------------------------------------------------------------------
template<int STAGE>
__global__ void __launch_bounds__(256) theta_stats_k(
    const float* __restrict__ coords, const int* __restrict__ nbr,
    const float* __restrict__ w1, const float* __restrict__ b1,
    const float* __restrict__ g1, const float* __restrict__ be1,
    const float* __restrict__ w2, const float* __restrict__ b2,
    const float* __restrict__ g2, const float* __restrict__ be2,
    const float* __restrict__ w3, const float* __restrict__ b3,
    const float* __restrict__ s1, const float* __restrict__ s2,
    float* __restrict__ sout, int rows, float inv_cnt)
{
    __shared__ float sw1[64], sb1[32], sw2[1024], sb2[32], sw3[512], sb3[16];
    __shared__ float n1s[32], n1t[32], n2s[32], n2t[32];
    __shared__ float acc[64];
    int tid = threadIdx.x;
    for (int i = tid; i < 64; i += 256) { sw1[i] = w1[i]; acc[i] = 0.f; }
    for (int i = tid; i < 32; i += 256) sb1[i] = b1[i];
    if (STAGE >= 2) {
        for (int i = tid; i < 1024; i += 256) sw2[i] = w2[i];
        for (int i = tid; i < 32; i += 256) {
            sb2[i] = b2[i];
            float m = s1[i]*inv_cnt;
            float v = s1[32+i]*inv_cnt - m*m;
            float sc = rsqrtf(v + EPS_) * g1[i];
            n1s[i] = sc; n1t[i] = be1[i] - m*sc;
        }
    }
    if (STAGE >= 3) {
        for (int i = tid; i < 512; i += 256) sw3[i] = w3[i];
        for (int i = tid; i < 16; i += 256) sb3[i] = b3[i];
        for (int i = tid; i < 32; i += 256) {
            float m = s2[i]*inv_cnt;
            float v = s2[32+i]*inv_cnt - m*m;
            float sc = rsqrtf(v + EPS_) * g2[i];
            n2s[i] = sc; n2t[i] = be2[i] - m*sc;
        }
    }
    __syncthreads();

    int r = blockIdx.x * 256 + tid;
    if (r < rows) {
        int p = r / KT_, j = r % KT_;
        int b = p / N_;
        int nb = nbr[p*KT_ + j];
        float qx = coords[p*2], qy = coords[p*2+1];
        float d0 = coords[(b*N_+nb)*2+0] - qx;
        float d1 = coords[(b*N_+nb)*2+1] - qy;
        float x1[32];
        #pragma unroll
        for (int c = 0; c < 32; ++c) x1[c] = d0*sw1[c] + d1*sw1[32+c] + sb1[c];
        if (STAGE == 1) {
            #pragma unroll
            for (int c = 0; c < 32; ++c) {
                atomicAdd(&acc[c], x1[c]);
                atomicAdd(&acc[32+c], x1[c]*x1[c]);
            }
        } else {
            float h1[32];
            #pragma unroll
            for (int c = 0; c < 32; ++c) h1[c] = siluf(x1[c]*n1s[c] + n1t[c]);
            float x2[32];
            #pragma unroll
            for (int c = 0; c < 32; ++c) {
                float s = sb2[c];
                #pragma unroll
                for (int i = 0; i < 32; ++i) s += h1[i]*sw2[i*32+c];
                x2[c] = s;
            }
            if (STAGE == 2) {
                #pragma unroll
                for (int c = 0; c < 32; ++c) {
                    atomicAdd(&acc[c], x2[c]);
                    atomicAdd(&acc[32+c], x2[c]*x2[c]);
                }
            } else {
                float h2[32];
                #pragma unroll
                for (int c = 0; c < 32; ++c) h2[c] = siluf(x2[c]*n2s[c] + n2t[c]);
                #pragma unroll
                for (int c = 0; c < 16; ++c) {
                    float s = sb3[c];
                    #pragma unroll
                    for (int i = 0; i < 32; ++i) s += h2[i]*sw3[i*16+c];
                    atomicAdd(&acc[c], s);
                    atomicAdd(&acc[32+c], s*s);
                }
            }
        }
    }
    __syncthreads();
    for (int i = tid; i < 64; i += 256) atomicAdd(&sout[i], acc[i]);
}

// ---------------------------------------------------------------------------
// Theta final pass: recompute full weightnet per neighbor, reduce over the
// 300 neighbors against density / signal channels (both share the same w).
// ---------------------------------------------------------------------------
__global__ void __launch_bounds__(256) theta_final_k(
    const float* __restrict__ coords, const int* __restrict__ nbr,
    const float* __restrict__ ctxv,
    const float* __restrict__ w1, const float* __restrict__ b1,
    const float* __restrict__ g1, const float* __restrict__ be1,
    const float* __restrict__ w2, const float* __restrict__ b2,
    const float* __restrict__ g2, const float* __restrict__ be2,
    const float* __restrict__ w3, const float* __restrict__ b3,
    const float* __restrict__ g3, const float* __restrict__ be3,
    const float* __restrict__ s1, const float* __restrict__ s2,
    const float* __restrict__ s3, float inv_cnt,
    float* __restrict__ dp_pc, float* __restrict__ sp_pc)
{
    __shared__ float sw1[64], sb1[32], sw2[1024], sb2[32], sw3[512], sb3[16];
    __shared__ float n1s[32], n1t[32], n2s[32], n2t[32], n3s[16], n3t[16];
    __shared__ float sdp[16], ssp[16];
    int tid = threadIdx.x;
    int p = blockIdx.x;
    int b = p / N_;
    for (int i = tid; i < 64; i += 256) sw1[i] = w1[i];
    for (int i = tid; i < 1024; i += 256) sw2[i] = w2[i];
    for (int i = tid; i < 512; i += 256) sw3[i] = w3[i];
    for (int i = tid; i < 32; i += 256) {
        sb1[i] = b1[i]; sb2[i] = b2[i];
        float m1 = s1[i]*inv_cnt, v1 = s1[32+i]*inv_cnt - m1*m1;
        float c1 = rsqrtf(v1 + EPS_) * g1[i];
        n1s[i] = c1; n1t[i] = be1[i] - m1*c1;
        float m2 = s2[i]*inv_cnt, v2 = s2[32+i]*inv_cnt - m2*m2;
        float c2 = rsqrtf(v2 + EPS_) * g2[i];
        n2s[i] = c2; n2t[i] = be2[i] - m2*c2;
    }
    for (int i = tid; i < 16; i += 256) {
        sb3[i] = b3[i];
        float m3 = s3[i]*inv_cnt, v3 = s3[32+i]*inv_cnt - m3*m3;
        float c3 = rsqrtf(v3 + EPS_) * g3[i];
        n3s[i] = c3; n3t[i] = be3[i] - m3*c3;
        sdp[i] = 0.f; ssp[i] = 0.f;
    }
    __syncthreads();

    float qx = coords[p*2], qy = coords[p*2+1];
    float ldp[16], lsp[16];
    #pragma unroll
    for (int m = 0; m < 16; ++m) { ldp[m] = 0.f; lsp[m] = 0.f; }

    for (int j = tid; j < KT_; j += 256) {
        int nb = nbr[p*KT_ + j];
        float d0 = coords[(b*N_+nb)*2+0] - qx;
        float d1 = coords[(b*N_+nb)*2+1] - qy;
        float h1[32];
        #pragma unroll
        for (int c = 0; c < 32; ++c) {
            float x = d0*sw1[c] + d1*sw1[32+c] + sb1[c];
            h1[c] = siluf(x*n1s[c] + n1t[c]);
        }
        float h2[32];
        #pragma unroll
        for (int c = 0; c < 32; ++c) {
            float s = sb2[c];
            #pragma unroll
            for (int i = 0; i < 32; ++i) s += h1[i]*sw2[i*32+c];
            h2[c] = siluf(s*n2s[c] + n2t[c]);
        }
        float dens = (nb < C_) ? 1.f : 0.f;
        float sig  = (nb < C_) ? ctxv[b*C_ + nb] : 0.f;
        #pragma unroll
        for (int c = 0; c < 16; ++c) {
            float s = sb3[c];
            #pragma unroll
            for (int i = 0; i < 32; ++i) s += h2[i]*sw3[i*16+c];
            float wv = siluf(s*n3s[c] + n3t[c]);
            ldp[c] += dens * wv;
            lsp[c] += sig  * wv;
        }
    }
    #pragma unroll
    for (int m = 0; m < 16; ++m) {
        atomicAdd(&sdp[m], ldp[m]);
        atomicAdd(&ssp[m], lsp[m]);
    }
    __syncthreads();
    if (tid < 16)      dp_pc[p*16 + tid] = sdp[tid];
    else if (tid < 32) sp_pc[p*16 + (tid-16)] = ssp[tid-16];
}

// ---------------------------------------------------------------------------
// WMMA GEMM:  C[M,N] = act( (A[M,K] * B[K,N] + bias) * scale )
// B is supplied TRANSPOSED (Bt: N x K row-major) so each lane's fragment is
// contiguous.  One wave per 16x16 C-tile, 4 waves per block.
// Branch-free operand path: row/col guards are address clamps, K-tail uses
// value selects (v_cndmask), never exec-mask branches.
// Alignment contract (satisfied by all call sites): if K >= 32 then K % 32 == 0
// and lda % 4 == 0; A and Bt are 256B-aligned workspace/param buffers.
// ---------------------------------------------------------------------------
__global__ void __launch_bounds__(128) gemm16_k(
    const float* __restrict__ A, int lda,
    const float* __restrict__ Bt,
    float* __restrict__ Cp, int ldc,
    int M, int K, int Nn,
    const float* __restrict__ bias, float scale, int relu)
{
    int lane = threadIdx.x;            // 0..31 (one wave per 16x16 tile)
    int m0 = (blockIdx.x * 4 + threadIdx.y) * 16;
    int n0 = blockIdx.y * 16;
    int ml = lane & 15, hi = lane >> 4;
    int arow = min(m0 + ml, M - 1);
    int bcol = min(n0 + ml, Nn - 1);
    const float* Ap = A  + (size_t)arow * lda;
    const float* Bp = Bt + (size_t)bcol * K;

    v8f acc = {};
    int Kfull = K & ~31;
    int kk = 0;
    #pragma unroll 2
    for (; kk < Kfull; kk += 32) {
        // A fragment: two contiguous 8-float runs per lane -> 4x b128
        const float4* a0 = (const float4*)(Ap + kk + hi*8);
        const float4* a1 = (const float4*)(Ap + kk + 16 + hi*8);
        float4 A0 = a0[0], A1 = a0[1], A2 = a1[0], A3 = a1[1];
        // B fragment: two contiguous 8-float runs per lane -> 4x b128
        const float4* b0 = (const float4*)(Bp + kk + hi*8);
        const float4* b1 = (const float4*)(Bp + kk + 16 + hi*8);
        float4 B0 = b0[0], B1 = b0[1], B2 = b1[0], B3 = b1[1];
        v16h av, bv;
        av[0]=(_Float16)A0.x; av[1]=(_Float16)A0.y; av[2]=(_Float16)A0.z; av[3]=(_Float16)A0.w;
        av[4]=(_Float16)A1.x; av[5]=(_Float16)A1.y; av[6]=(_Float16)A1.z; av[7]=(_Float16)A1.w;
        av[8]=(_Float16)A2.x; av[9]=(_Float16)A2.y; av[10]=(_Float16)A2.z; av[11]=(_Float16)A2.w;
        av[12]=(_Float16)A3.x; av[13]=(_Float16)A3.y; av[14]=(_Float16)A3.z; av[15]=(_Float16)A3.w;
        bv[0]=(_Float16)B0.x; bv[1]=(_Float16)B0.y; bv[2]=(_Float16)B0.z; bv[3]=(_Float16)B0.w;
        bv[4]=(_Float16)B1.x; bv[5]=(_Float16)B1.y; bv[6]=(_Float16)B1.z; bv[7]=(_Float16)B1.w;
        bv[8]=(_Float16)B2.x; bv[9]=(_Float16)B2.y; bv[10]=(_Float16)B2.z; bv[11]=(_Float16)B2.w;
        bv[12]=(_Float16)B3.x; bv[13]=(_Float16)B3.y; bv[14]=(_Float16)B3.z; bv[15]=(_Float16)B3.w;
        acc = __builtin_amdgcn_wmma_f32_16x16x32_f16(false, av, false, bv,
                                                     (short)0, acc, false, false);
    }
    if (kk < K) {   // tail (only K=2 / K=16 call sites reach this)
        v16h av, bv;
        #pragma unroll
        for (int e = 0; e < 16; ++e) {
            int ka = kk + ((e < 8) ? (hi*8 + e) : (16 + hi*8 + (e - 8)));
            float fa = Ap[min(ka, K - 1)];
            av[e] = (_Float16)((ka < K) ? fa : 0.f);
            int kb = kk + ((e < 8) ? (hi*8 + e) : (16 + hi*8 + (e - 8)));
            float fb = Bp[min(kb, K - 1)];
            bv[e] = (_Float16)((kb < K) ? fb : 0.f);
        }
        acc = __builtin_amdgcn_wmma_f32_16x16x32_f16(false, av, false, bv,
                                                     (short)0, acc, false, false);
    }
    #pragma unroll
    for (int r = 0; r < 8; ++r) {
        int row = m0 + r + 8*hi;
        int col = n0 + ml;
        if (row < M && col < Nn) {
            float v = acc[r];
            if (bias) v += bias[col];
            v *= scale;
            if (relu) v = fmaxf(v, 0.f);
            Cp[(size_t)row*ldc + col] = v;
        }
    }
}

// ---------------------------------------------------------------------------
// BN statistics over all rows (per-channel sum / sumsq), LDS ds_add_f32 + 1
// global atomic per channel per block.
// ---------------------------------------------------------------------------
__global__ void __launch_bounds__(256) bn_stats_k(const float* __restrict__ x,
                                                  int n, int Cc, float* __restrict__ stats) {
    __shared__ float s[64];
    int tid = threadIdx.x;
    if (tid < 64) s[tid] = 0.f;
    __syncthreads();
    int i = blockIdx.x * 256 + tid;
    if (i < n) {
        int c = i % Cc;
        float v = x[i];
        atomicAdd(&s[c], v);
        atomicAdd(&s[32+c], v*v);
    }
    __syncthreads();
    if (tid < 64) atomicAdd(&stats[tid], s[tid]);
}

__global__ void __launch_bounds__(256) bn_apply_silu_k(
    float* __restrict__ x, int n, int Cc,
    const float* __restrict__ stats,
    const float* __restrict__ g, const float* __restrict__ be, float inv_cnt) {
    int i = blockIdx.x * 256 + threadIdx.x;
    if (i >= n) return;
    int c = i % Cc;
    float m = stats[c]*inv_cnt;
    float v = stats[32+c]*inv_cnt - m*m;
    float sc = rsqrtf(v + EPS_) * g[c];
    float xn = (x[i] - m)*sc + be[c];
    x[i] = siluf(xn);
}

// ---------------------------------------------------------------------------
// deltas for the 5-NN convs (first 5 entries of sorted 300-NN list)
// ---------------------------------------------------------------------------
__global__ void build_deltas5_k(const float* __restrict__ coords,
                                const int* __restrict__ nbr,
                                float* __restrict__ deltas5) {
    int i = blockIdx.x * blockDim.x + threadIdx.x;
    if (i >= R5_) return;
    int p = i / KC_, j = i % KC_;
    int b = p / N_;
    int nb = nbr[p*KT_ + j];
    deltas5[i*2+0] = coords[(b*N_+nb)*2+0] - coords[p*2+0];
    deltas5[i*2+1] = coords[(b*N_+nb)*2+1] - coords[p*2+1];
}

// ---------------------------------------------------------------------------
// einsum bnkc,bnkm->bncm for k=5:  pc[p, c*16+m] = sum_j hin[nb_j, c]*w5[p,j,m]
// ---------------------------------------------------------------------------
__global__ void __launch_bounds__(256) pc_einsum_k(
    const float* __restrict__ hin, int cin,
    const float* __restrict__ w5, const int* __restrict__ nbr,
    float* __restrict__ pc, int total) {
    int gid = blockIdx.x * 256 + threadIdx.x;
    if (gid >= total) return;
    int p = gid / cin, c = gid % cin;
    int b = p / N_;
    const int* id = &nbr[p*KT_];
    float acc[16];
    #pragma unroll
    for (int m = 0; m < 16; ++m) acc[m] = 0.f;
    #pragma unroll
    for (int j = 0; j < KC_; ++j) {
        int nb = id[j];
        float v = hin[(size_t)(b*N_ + nb)*cin + c];
        const float* wr = &w5[(size_t)(p*KC_ + j)*16];
        #pragma unroll
        for (int m = 0; m < 16; ++m) acc[m] += v * wr[m];
    }
    float* o = &pc[(size_t)p*cin*16 + (size_t)c*16];
    #pragma unroll
    for (int m = 0; m < 16; ++m) o[m] = acc[m];
}

// ---------------------------------------------------------------------------
// Finalize: d_out = [ f_mu (B*T) | sigma (B*T*T diag softplus) ]
// ---------------------------------------------------------------------------
__global__ void finalize_k(const float* __restrict__ f4, float* __restrict__ out, int total) {
    int i = blockIdx.x * blockDim.x + threadIdx.x;
    if (i >= total) return;
    if (i < B_*T_) {
        int b = i / T_, t = i % T_;
        out[i] = f4[((size_t)(b*N_) + C_ + t)*2 + 0];
    } else {
        int r = i - B_*T_;
        int b = r / (T_*T_);
        int rr = r % (T_*T_);
        int t = rr / T_, t2 = rr % T_;
        float v = 0.f;
        if (t == t2) {
            float x = f4[((size_t)(b*N_) + C_ + t)*2 + 1];
            v = (x > 20.f) ? x : log1pf(__expf(x));
        }
        out[i] = v;
    }
}

// ===========================================================================
// Host launcher
// ===========================================================================
extern "C" void kernel_launch(void* const* d_in, const int* in_sizes, int n_in,
                              void* d_out, int out_size, void* d_ws, size_t ws_size,
                              hipStream_t stream) {
    (void)in_sizes; (void)n_in; (void)out_size;
    const float* ctx_coords = (const float*)d_in[0];
    const float* ctx_values = (const float*)d_in[1];
    const float* tgt_coords = (const float*)d_in[2];

    // params pytree flattened with sorted dict keys:
    // blocks: c1,c2,c3,c4,theta ; per block: Wl,b1,b2,b3,be1,be2,be3,bl,g1,g2,g3,w1,w2,w3
    enum { iWl = 0, ib1, ib2, ib3, ibe1, ibe2, ibe3, ibl, ig1, ig2, ig3, iw1, iw2, iw3 };
    enum { Bc1 = 0, Bc2, Bc3, Bc4, Bth };
    auto P = [&](int blk, int item) { return (const float*)d_in[3 + blk*14 + item]; };

    // ---- workspace layout ----
    char* wsb = (char*)d_ws;
    size_t off = 0;
    auto alloc = [&](size_t bytes) { size_t o = off; off = (off + bytes + 255) & ~(size_t)255; return o; };
    float* coords  = (float*)(wsb + alloc((size_t)BN_*2*4));
    int*   nbr     = (int*)  (wsb + alloc((size_t)BN_*KT_*4));
    float* statsT1 = (float*)(wsb + alloc(64*4));
    float* statsT2 = (float*)(wsb + alloc(64*4));
    float* statsT3 = (float*)(wsb + alloc(64*4));
    float* statsC  = (float*)(wsb + alloc(64*4));
    float* dp_pc   = (float*)(wsb + alloc((size_t)BN_*16*4));
    float* sp_pc   = (float*)(wsb + alloc((size_t)BN_*16*4));
    float* h256    = (float*)(wsb + alloc((size_t)BN_*256*4));
    float* deltas5 = (float*)(wsb + alloc((size_t)R5_*2*4));
    float* x5a     = (float*)(wsb + alloc((size_t)R5_*32*4));
    float* x5b     = (float*)(wsb + alloc((size_t)R5_*32*4));
    float* w5      = (float*)(wsb + alloc((size_t)R5_*16*4));
    float* pcbuf   = (float*)(wsb + alloc((size_t)BN_*256*16*4));
    float* btbuf   = (float*)(wsb + alloc((size_t)4096*16*4));   // transposed-B scratch (max K*N)
    float* h1c     = (float*)(wsb + alloc((size_t)BN_*16*4));
    float* h2c     = (float*)(wsb + alloc((size_t)BN_*32*4));
    float* h3c     = (float*)(wsb + alloc((size_t)BN_*16*4));
    float* f4      = (float*)(wsb + alloc((size_t)BN_*2*4));
    if (off > ws_size) return;   // workspace too small — bail deterministically

    const float invT = 1.f / (float)RT_;
    const float inv5 = 1.f / (float)R5_;

    // Transpose B (K x N -> N x K) into btbuf, then tiled WMMA GEMM.
    auto gemm = [&](const float* A, int lda, const float* Bw,
                    float* Cp, int ldc, int M, int K, int Nn,
                    const float* bias, float scale, int relu) {
        int tn = K * Nn;
        transpose_k<<<(tn + 255) / 256, 256, 0, stream>>>(Bw, btbuf, K, Nn);
        dim3 g((M + 63) / 64, (Nn + 15) / 16);
        gemm16_k<<<g, dim3(32, 4), 0, stream>>>(A, lda, btbuf, Cp, ldc, M, K, Nn,
                                                bias, scale, relu);
    };

    // ---- coords + neighbor lists ----
    build_coords_k<<<(BN_ + 255) / 256, 256, 0, stream>>>(ctx_coords, tgt_coords, coords);
    topk_k<<<BN_, 256, 0, stream>>>(coords, nbr);

    // ---- theta weightnet: 3 stats passes (recompute) + fused final reduce ----
    zero_k<<<1, 64, 0, stream>>>(statsT1, 64);
    zero_k<<<1, 64, 0, stream>>>(statsT2, 64);
    zero_k<<<1, 64, 0, stream>>>(statsT3, 64);
    const int gT = (RT_ + 255) / 256;
    theta_stats_k<1><<<gT, 256, 0, stream>>>(coords, nbr,
        P(Bth,iw1), P(Bth,ib1), P(Bth,ig1), P(Bth,ibe1),
        P(Bth,iw2), P(Bth,ib2), P(Bth,ig2), P(Bth,ibe2),
        P(Bth,iw3), P(Bth,ib3), statsT1, statsT2, statsT1, RT_, invT);
    theta_stats_k<2><<<gT, 256, 0, stream>>>(coords, nbr,
        P(Bth,iw1), P(Bth,ib1), P(Bth,ig1), P(Bth,ibe1),
        P(Bth,iw2), P(Bth,ib2), P(Bth,ig2), P(Bth,ibe2),
        P(Bth,iw3), P(Bth,ib3), statsT1, statsT2, statsT2, RT_, invT);
    theta_stats_k<3><<<gT, 256, 0, stream>>>(coords, nbr,
        P(Bth,iw1), P(Bth,ib1), P(Bth,ig1), P(Bth,ibe1),
        P(Bth,iw2), P(Bth,ib2), P(Bth,ig2), P(Bth,ibe2),
        P(Bth,iw3), P(Bth,ib3), statsT1, statsT2, statsT3, RT_, invT);
    theta_final_k<<<BN_, 256, 0, stream>>>(coords, nbr, ctx_values,
        P(Bth,iw1), P(Bth,ib1), P(Bth,ig1), P(Bth,ibe1),
        P(Bth,iw2), P(Bth,ib2), P(Bth,ig2), P(Bth,ibe2),
        P(Bth,iw3), P(Bth,ib3), P(Bth,ig3), P(Bth,ibe3),
        statsT1, statsT2, statsT3, invT, dp_pc, sp_pc);

    // Wl projections for dp / sp (WMMA GEMM), writing into concatenated h256
    gemm(dp_pc, 16, P(Bth,iWl), h256 + 0,   256, BN_, 16, 128, P(Bth,ibl), 1.f/(float)KT_, 0);
    gemm(sp_pc, 16, P(Bth,iWl), h256 + 128, 256, BN_, 16, 128, P(Bth,ibl), 1.f/(float)KT_, 0);

    // ---- cnn conv layers (k=5): weightnet fully via WMMA GEMMs ----
    build_deltas5_k<<<(R5_ + 255) / 256, 256, 0, stream>>>(coords, nbr, deltas5);

    auto conv_layer = [&](int blk, const float* hin, int cin, float* outp, int cout, int relu) {
        // weightnet layer 1: (R5 x 2) @ (2 x 32)
        zero_k<<<1, 64, 0, stream>>>(statsC, 64);
        gemm(deltas5, 2, P(blk,iw1), x5a, 32, R5_, 2, 32, P(blk,ib1), 1.f, 0);
        bn_stats_k<<<(R5_*32 + 255) / 256, 256, 0, stream>>>(x5a, R5_*32, 32, statsC);
        bn_apply_silu_k<<<(R5_*32 + 255) / 256, 256, 0, stream>>>(x5a, R5_*32, 32, statsC,
            P(blk,ig1), P(blk,ibe1), inv5);
        // weightnet layer 2: (R5 x 32) @ (32 x 32)
        zero_k<<<1, 64, 0, stream>>>(statsC, 64);
        gemm(x5a, 32, P(blk,iw2), x5b, 32, R5_, 32, 32, P(blk,ib2), 1.f, 0);
        bn_stats_k<<<(R5_*32 + 255) / 256, 256, 0, stream>>>(x5b, R5_*32, 32, statsC);
        bn_apply_silu_k<<<(R5_*32 + 255) / 256, 256, 0, stream>>>(x5b, R5_*32, 32, statsC,
            P(blk,ig2), P(blk,ibe2), inv5);
        // weightnet layer 3: (R5 x 32) @ (32 x 16)
        zero_k<<<1, 64, 0, stream>>>(statsC, 64);
        gemm(x5b, 32, P(blk,iw3), w5, 16, R5_, 32, 16, P(blk,ib3), 1.f, 0);
        bn_stats_k<<<(R5_*16 + 255) / 256, 256, 0, stream>>>(w5, R5_*16, 16, statsC);
        bn_apply_silu_k<<<(R5_*16 + 255) / 256, 256, 0, stream>>>(w5, R5_*16, 16, statsC,
            P(blk,ig3), P(blk,ibe3), inv5);
        // einsum + Wl projection (WMMA GEMM, K = cin*16)
        int total = BN_ * cin;
        pc_einsum_k<<<(total + 255) / 256, 256, 0, stream>>>(hin, cin, w5, nbr, pcbuf, total);
        gemm(pcbuf, cin*16, P(blk,iWl), outp, cout, BN_, cin*16, cout,
             P(blk,ibl), 1.f/(float)KC_, relu);
    };

    conv_layer(Bc1, h256, 256, h1c, 16, 1);
    conv_layer(Bc2, h1c,  16,  h2c, 32, 1);
    conv_layer(Bc3, h2c,  32,  h3c, 16, 1);
    conv_layer(Bc4, h3c,  16,  f4,  2,  0);

    // ---- outputs: mu + diagonal sigma ----
    const int total_out = B_*T_ + B_*T_*T_;
    finalize_k<<<(total_out + 255) / 256, 256, 0, stream>>>(f4, (float*)d_out, total_out);
}
